// GraphTransformer_7567732375848
// MI455X (gfx1250) — matrix operand order repaired and checked by reading, not compiled, not run on hardware
//
#include <hip/hip_runtime.h>
#include <hip/hip_bf16.h>
#include <math.h>

// ---------------- problem constants (from reference) ----------------
#define NN 50000
#define EE 800000
#define IN_DIM 128
#define HID 96
#define OUT_DIM 96
#define NHEAD 4
#define CH 24          // channels per head
#define QKVS 384       // 4 * 96 fused output columns (Q|K|V|S)
#define LN_EPS 1e-5f

typedef __attribute__((ext_vector_type(16))) _Float16 v16h;
typedef __attribute__((ext_vector_type(8)))  _Float16 v8h;
typedef __attribute__((ext_vector_type(8)))  float    v8f;

// ---------------- small helpers ----------------
__device__ __forceinline__ unsigned f2ord(float f) {
    unsigned u = __float_as_uint(f);
    return (u & 0x80000000u) ? ~u : (u | 0x80000000u);
}
__device__ __forceinline__ float ord2f(unsigned u) {
    unsigned b = (u & 0x80000000u) ? (u & 0x7FFFFFFFu) : ~u;
    return __uint_as_float(b);
}
__device__ __forceinline__ float wave_sum32(float v) {
    #pragma unroll
    for (int off = 16; off > 0; off >>= 1) v += __shfl_xor(v, off, 32);
    return v;
}

// ---------------- fp32 -> fp16 conversion ----------------
__global__ __launch_bounds__(256) void f32_to_f16_kernel(
    const float* __restrict__ in, _Float16* __restrict__ out, int n) {
    int i = blockIdx.x * 256 + threadIdx.x;
    if (i < n) out[i] = (_Float16)in[i];
}

// ---------------- pack 4 weight mats (+bias) into fused [384,K] f16 ----------------
__global__ __launch_bounds__(256) void pack_w_kernel(
    const float* __restrict__ Wq, const float* __restrict__ Wk,
    const float* __restrict__ Wv, const float* __restrict__ Ws,
    const float* __restrict__ bq, const float* __restrict__ bk,
    const float* __restrict__ bv, const float* __restrict__ bs,
    _Float16* __restrict__ Wcat, float* __restrict__ bcat, int K) {
    int idx = blockIdx.x * 256 + threadIdx.x;
    int total = QKVS * K;
    if (idx >= total) return;
    int r = idx / K, c = idx - r * K;
    int seg = r / HID, rr = r - seg * HID;
    const float* W = (seg == 0) ? Wq : (seg == 1) ? Wk : (seg == 2) ? Wv : Ws;
    Wcat[idx] = (_Float16)W[rr * K + c];
    if (c == 0) {
        const float* b = (seg == 0) ? bq : (seg == 1) ? bk : (seg == 2) ? bv : bs;
        bcat[r] = b[rr];
    }
}

// ---------------- fused QKVS projection: out[M,384] = Xh[M,K] @ Wh[384,K]^T + b ----
// 192 threads = 6 waves; each wave owns a 16x64 strip (4 accumulators), so one A
// fragment per k-step feeds 4 back-to-back v_wmma_f32_16x16x32_f16 (hides the
// WMMA->WMMA hazard and cuts A loads 4x). grid = (M/16, 1).
template <int K>
__global__ __launch_bounds__(192) void gemm_qkvs_kernel(
    const _Float16* __restrict__ Xh,    // [M, K] row-major
    const _Float16* __restrict__ Wh,    // [384, K] row-major (col j of B = row j of W)
    const float*    __restrict__ bcat,  // [384]
    float* __restrict__ out,            // [M, 384]
    int M) {
    const int lane = threadIdx.x & 31;
    const int wave = threadIdx.x >> 5;        // 0..5
    const int m0   = blockIdx.x * 16;
    const int col0 = wave * 64;               // wave strip: 64 output columns
    if (m0 >= M) return;

    const int n   = lane & 15;   // output column within tile / A row within tile
    const int hiL = lane >> 4;   // lane group 0/1

    // C/D layout: element r of lane -> (row m0 + r + 8*hiL, col col0 + j*16 + n)
    v8f acc[4];
    #pragma unroll
    for (int j = 0; j < 4; ++j) {
        const float bval = bcat[col0 + j * 16 + n];
        #pragma unroll
        for (int r = 0; r < 8; ++r) acc[j][r] = bval;
    }

    const int arow = (m0 + n < M) ? (m0 + n) : (M - 1);
    const _Float16* xrow = Xh + (size_t)arow * K;
    const _Float16* wrow0 = Wh + (size_t)(col0 + n) * K;   // tile j adds j*16*K

    #pragma unroll
    for (int kk = 0; kk < K; kk += 32) {
        // A fragment (16x32 f16): lanes 0-15 hold K{kk..kk+15},
        // lanes 16-31 hold K{kk+16..kk+31} of row n.
        const _Float16* ap = xrow + kk + hiL * 16;
        v16h afull = *(const v16h*)(ap);
        v8h alo = afull.lo;
        v8h ahi = afull.hi;
        v16h a;
        #pragma unroll
        for (int i = 0; i < 8; ++i) { a[i] = alo[i]; a[8 + i] = ahi[i]; }

        if (kk + 32 < K) __builtin_prefetch(xrow + kk + 32, 0, 3);

        // 4 B fragments (32x16 f16): lane n holds 16 contiguous K-halves of its
        // column; lane group selects K half [kk..kk+15] vs [kk+16..kk+31].
        #pragma unroll
        for (int j = 0; j < 4; ++j) {
            v16h b = *(const v16h*)(wrow0 + (size_t)j * 16 * K + kk + hiL * 16);
            acc[j] = __builtin_amdgcn_wmma_f32_16x16x32_f16(
                /*neg_a=*/false, a, /*neg_b=*/false, b,
                /*c_mod=*/(short)0, acc[j], /*reuse_a=*/false, /*reuse_b=*/false);
        }
    }

    float* obase = out + (size_t)m0 * QKVS + col0 + n;
    if (m0 + 16 <= M) {
        // full tile: branch-free strided stores
        #pragma unroll
        for (int j = 0; j < 4; ++j) {
            float* op = obase + j * 16;
            #pragma unroll
            for (int r = 0; r < 8; ++r)
                op[(size_t)(r + 8 * hiL) * QKVS] = acc[j][r];
        }
    } else {
        #pragma unroll
        for (int j = 0; j < 4; ++j) {
            float* op = obase + j * 16;
            #pragma unroll
            for (int r = 0; r < 8; ++r) {
                int row = r + 8 * hiL;
                if (m0 + row < M) op[(size_t)row * QKVS] = acc[j][r];
            }
        }
    }
}

// ---------------- attention scratch init ----------------
__global__ __launch_bounds__(256) void init_attn_kernel(
    unsigned* __restrict__ amax, float* __restrict__ denom,
    float* __restrict__ agg, int nAmax, int nAgg) {
    int i = blockIdx.x * 256 + threadIdx.x;
    if (i < nAmax) { amax[i] = f2ord(-INFINITY); denom[i] = 0.f; }
    if (i < nAgg)  agg[i] = 0.f;
}

// ---------------- per-edge logits + segment max (thread = edge*head) -------------
__global__ __launch_bounds__(256) void edge_logits_kernel(
    const int* __restrict__ srcs, const int* __restrict__ dsts,
    const float* __restrict__ qkvs, float* __restrict__ alpha,
    unsigned* __restrict__ amax, int E_) {
    int t = blockIdx.x * 256 + threadIdx.x;
    if (t >= E_ * NHEAD) return;
    int e = t >> 2, h = t & 3;
    int s = srcs[e], d = dsts[e];
    const float* q = qkvs + (size_t)d * QKVS + h * CH;           // Q block [0,96)
    const float* k = qkvs + (size_t)s * QKVS + HID + h * CH;     // K block [96,192)
    float acc = 0.f;
    #pragma unroll
    for (int c = 0; c < CH; ++c) acc += q[c] * k[c];
    acc *= 0.2041241452319315f;   // 1/sqrt(24)
    alpha[t] = acc;
    atomicMax(amax + (size_t)d * NHEAD + h, f2ord(acc));
}

// ---------------- per-edge exp + denom + unnormalized message scatter ------------
__global__ __launch_bounds__(256) void edge_msg_kernel(
    const int* __restrict__ srcs, const int* __restrict__ dsts,
    const float* __restrict__ qkvs, const float* __restrict__ alpha,
    const unsigned* __restrict__ amax, float* __restrict__ denom,
    float* __restrict__ agg, int E_) {
    int t = blockIdx.x * 256 + threadIdx.x;
    if (t >= E_ * NHEAD) return;
    int e = t >> 2, h = t & 3;
    int s = srcs[e], d = dsts[e];
    float a = expf(alpha[t] - ord2f(amax[(size_t)d * NHEAD + h]));
    atomicAdd(denom + (size_t)d * NHEAD + h, a);
    const float* v = qkvs + (size_t)s * QKVS + 2 * HID + h * CH; // V block [192,288)
    float* o = agg + (size_t)d * HID + h * CH;
    #pragma unroll
    for (int c = 0; c < CH; ++c) atomicAdd(o + c, a * v[c]);
}

// ---------------- per-node gate (+ optional GELU + f16 copy); 1 wave per node ----
template <bool GELU>
__global__ __launch_bounds__(256) void node_gate_kernel(
    const float* __restrict__ qkvs, const float* __restrict__ agg,
    const float* __restrict__ denom, const float* __restrict__ Wb,  // [288]
    float* __restrict__ hout, _Float16* __restrict__ hout16, int M) {
    int node = blockIdx.x * 8 + (threadIdx.x >> 5);
    if (node >= M) return;
    int lane = threadIdx.x & 31;
    float o[3], xr[3];
    float part = 0.f;
    #pragma unroll
    for (int j = 0; j < 3; ++j) {
        int c = lane + 32 * j;
        int h = c / CH;
        float dn = denom[(size_t)node * NHEAD + h];
        float ov = agg[(size_t)node * HID + c];
        ov = (dn > 0.f) ? (ov / dn) : 0.f;               // empty segment -> 0
        float xv = qkvs[(size_t)node * QKVS + 3 * HID + c]; // S block [288,384)
        o[j] = ov; xr[j] = xv;
        part += ov * Wb[c] + xv * Wb[HID + c] + (ov - xv) * Wb[2 * HID + c];
    }
    part = wave_sum32(part);
    float g = 1.f / (1.f + expf(-part));
    #pragma unroll
    for (int j = 0; j < 3; ++j) {
        int c = lane + 32 * j;
        float hv = g * xr[j] + (1.f - g) * o[j];
        if (GELU) hv = 0.5f * hv * (1.f + erff(hv * 0.70710678118654752f));
        hout[(size_t)node * HID + c] = hv;
        if (hout16) hout16[(size_t)node * HID + c] = (_Float16)hv;
    }
}

// ---------------- LayerNorm over 96 channels; 1 wave per node --------------------
__global__ __launch_bounds__(256) void layernorm_kernel(
    const float* __restrict__ h, const float* __restrict__ gamma,
    const float* __restrict__ beta, float* __restrict__ out, int M) {
    int node = blockIdx.x * 8 + (threadIdx.x >> 5);
    if (node >= M) return;
    int lane = threadIdx.x & 31;
    const float* hr = h + (size_t)node * HID;
    float v0 = hr[lane], v1 = hr[lane + 32], v2 = hr[lane + 64];
    float mu = wave_sum32(v0 + v1 + v2) * (1.f / 96.f);
    float d0 = v0 - mu, d1 = v1 - mu, d2 = v2 - mu;
    float var = wave_sum32(d0 * d0 + d1 * d1 + d2 * d2) * (1.f / 96.f);
    float inv = rsqrtf(var + LN_EPS);
    float* orow = out + (size_t)node * HID;
    orow[lane]      = d0 * inv * gamma[lane]      + beta[lane];
    orow[lane + 32] = d1 * inv * gamma[lane + 32] + beta[lane + 32];
    orow[lane + 64] = d2 * inv * gamma[lane + 64] + beta[lane + 64];
}

// ---------------- host side ----------------
extern "C" void kernel_launch(void* const* d_in, const int* in_sizes, int n_in,
                              void* d_out, int out_size, void* d_ws, size_t ws_size,
                              hipStream_t stream) {
    const float* x   = (const float*)d_in[0];
    const int*   ei  = (const int*)d_in[1];
    const int M = in_sizes[0] / IN_DIM;   // 50000
    const int E = in_sizes[1] / 2;        // 800000
    const int* srcs = ei;
    const int* dsts = ei + E;

    const float *Wq1 = (const float*)d_in[2],  *bq1 = (const float*)d_in[3];
    const float *Wk1 = (const float*)d_in[4],  *bk1 = (const float*)d_in[5];
    const float *Wv1 = (const float*)d_in[6],  *bv1 = (const float*)d_in[7];
    const float *Ws1 = (const float*)d_in[8],  *bs1 = (const float*)d_in[9];
    const float *Wb1 = (const float*)d_in[10];
    const float *Wq2 = (const float*)d_in[11], *bq2 = (const float*)d_in[12];
    const float *Wk2 = (const float*)d_in[13], *bk2 = (const float*)d_in[14];
    const float *Wv2 = (const float*)d_in[15], *bv2 = (const float*)d_in[16];
    const float *Ws2 = (const float*)d_in[17], *bs2 = (const float*)d_in[18];
    const float *Wb2 = (const float*)d_in[19];
    const float *gamma = (const float*)d_in[20], *beta = (const float*)d_in[21];

    // workspace layout (all offsets 256B-aligned)
    char* ws = (char*)d_ws;
    size_t off = 0;
    auto take = [&](size_t bytes) { char* p = ws + off; off += (bytes + 255) & ~(size_t)255; return p; };
    _Float16* xh    = (_Float16*)take((size_t)M * IN_DIM * 2);   // 12.8 MB
    _Float16* W1h   = (_Float16*)take((size_t)QKVS * IN_DIM * 2);
    _Float16* W2h   = (_Float16*)take((size_t)QKVS * HID * 2);
    float*    b1c   = (float*)   take(QKVS * 4);
    float*    b2c   = (float*)   take(QKVS * 4);
    float*    qkvs  = (float*)   take((size_t)M * QKVS * 4);     // 76.8 MB (reused)
    float*    alpha = (float*)   take((size_t)E * NHEAD * 4);    // 12.8 MB (reused)
    unsigned* amax  = (unsigned*)take((size_t)M * NHEAD * 4);
    float*    denom = (float*)   take((size_t)M * NHEAD * 4);
    float*    agg   = (float*)   take((size_t)M * HID * 4);      // 19.2 MB (reused)
    float*    hbuf  = (float*)   take((size_t)M * HID * 4);      // 19.2 MB (reused)
    _Float16* hh    = (_Float16*)take((size_t)M * HID * 2);      // 9.6 MB
    (void)ws_size; (void)n_in; (void)out_size;

    const int THR = 256;
    const int GEMM_THR = 192;                 // 6 waves x 16x64 strip = 16x384
    int gemmBlocks = (M + 15) / 16;
    int nodeBlocks = (M + 7) / 8;
    int ehThreads  = E * NHEAD;

    // ---- setup: f16 activations + fused weights ----
    f32_to_f16_kernel<<<(M * IN_DIM + THR - 1) / THR, THR, 0, stream>>>(x, xh, M * IN_DIM);
    pack_w_kernel<<<(QKVS * IN_DIM + THR - 1) / THR, THR, 0, stream>>>(
        Wq1, Wk1, Wv1, Ws1, bq1, bk1, bv1, bs1, W1h, b1c, IN_DIM);
    pack_w_kernel<<<(QKVS * HID + THR - 1) / THR, THR, 0, stream>>>(
        Wq2, Wk2, Wv2, Ws2, bq2, bk2, bv2, bs2, W2h, b2c, HID);

    // ---- layer 1 ----
    gemm_qkvs_kernel<IN_DIM><<<gemmBlocks, GEMM_THR, 0, stream>>>(xh, W1h, b1c, qkvs, M);
    init_attn_kernel<<<(M * HID + THR - 1) / THR, THR, 0, stream>>>(
        amax, denom, agg, M * NHEAD, M * HID);
    edge_logits_kernel<<<(ehThreads + THR - 1) / THR, THR, 0, stream>>>(
        srcs, dsts, qkvs, alpha, amax, E);
    edge_msg_kernel<<<(ehThreads + THR - 1) / THR, THR, 0, stream>>>(
        srcs, dsts, qkvs, alpha, amax, denom, agg, E);
    node_gate_kernel<true><<<nodeBlocks, THR, 0, stream>>>(
        qkvs, agg, denom, Wb1, hbuf, hh, M);

    // ---- layer 2 ----
    gemm_qkvs_kernel<HID><<<gemmBlocks, GEMM_THR, 0, stream>>>(hh, W2h, b2c, qkvs, M);
    init_attn_kernel<<<(M * HID + THR - 1) / THR, THR, 0, stream>>>(
        amax, denom, agg, M * NHEAD, M * HID);
    edge_logits_kernel<<<(ehThreads + THR - 1) / THR, THR, 0, stream>>>(
        srcs, dsts, qkvs, alpha, amax, E);
    edge_msg_kernel<<<(ehThreads + THR - 1) / THR, THR, 0, stream>>>(
        srcs, dsts, qkvs, alpha, amax, denom, agg, E);
    node_gate_kernel<false><<<nodeBlocks, THR, 0, stream>>>(
        qkvs, agg, denom, Wb2, hbuf, (_Float16*)nullptr, M);

    // ---- layernorm -> output ----
    layernorm_kernel<<<nodeBlocks, THR, 0, stream>>>(hbuf, gamma, beta, (float*)d_out, M);
}
